// SetConvDecoder_29472065585208
// MI455X (gfx1250) — compile-verified, matrix operand order checked by reading, not run on hardware
//
#include <hip/hip_runtime.h>

// ---------------------------------------------------------------------------
// SetConvDecoder fused RBF-weights + GEMM for MI455X (gfx1250, wave32, WMMA)
//
// out[b,t,z*NK+k] = sum_g exp(-0.5*sum_d ((xt[b,t,d]-xg[b,g,d])/ls[d,k])^2) * z[b,g,z]
//
// Strategy: weights computed in fp32 (v_exp_f32), rounded to f16; Z pre-cast/
// transposed to f16 (b,z,g) in d_ws; accumulate fp32 with
// v_wmma_f32_16x16x32_f16. One wave = 16(t) x 128(z) strip of one (b,k).
// ---------------------------------------------------------------------------

typedef __attribute__((ext_vector_type(16))) _Float16 v16h;
typedef __attribute__((ext_vector_type(8)))  _Float16 v8h;
typedef __attribute__((ext_vector_type(8)))  float    v8f;
typedef __attribute__((ext_vector_type(4)))  float    v4f;

enum : int { NB = 4, NG = 4096, NTT = 2048, DXX = 2, DZZ = 128, NKK = 2 };

// ---------------------------------------------------------------------------
// Kernel 1: z_grid (b,g,z) f32  ->  zt (b,z,g) f16, LDS-tiled 32x32 transpose.
// Reads coalesced along z, writes coalesced along g.
// ---------------------------------------------------------------------------
__global__ __launch_bounds__(256) void transpose_cast_z(
    const float* __restrict__ z, _Float16* __restrict__ zt) {
  __shared__ float tile[32][33];  // 33 pad -> conflict-free column reads

  const int bi  = blockIdx.x;          // NB * (NG/32) * (DZZ/32) = 2048 blocks
  const int b   = bi >> 9;
  const int rem = bi & 511;
  const int g0  = (rem >> 2) << 5;     // 128 g-tiles
  const int z0  = (rem & 3) << 5;      // 4 z-tiles

  const int zl = threadIdx.x & 31;
  const int gq = threadIdx.x >> 5;     // 0..7

#pragma unroll
  for (int j = 0; j < 4; ++j) {
    const int gl = gq + 8 * j;
    tile[gl][zl] = z[((b * NG + g0 + gl) * DZZ) + z0 + zl];
  }
  __syncthreads();

  const int gl2 = threadIdx.x & 31;
  const int zq  = threadIdx.x >> 5;
#pragma unroll
  for (int j = 0; j < 4; ++j) {
    const int zl2 = zq + 8 * j;
    zt[((b * DZZ + z0 + zl2) * NG) + g0 + gl2] = (_Float16)tile[gl2][zl2];
  }
}

// ---------------------------------------------------------------------------
// Kernel 2: fused weight-gen + WMMA GEMM.
// Grid: 256 blocks x 128 threads (4 waves). Wave -> one (b,k, 16-row M tile).
// ---------------------------------------------------------------------------
__global__ __launch_bounds__(128) void setconv_wmma(
    const float* __restrict__ xg,    // (NB, NG, 2)
    const float* __restrict__ xt,    // (NB, NTT, 2)
    const float* __restrict__ lsp,   // (2, NKK)
    const _Float16* __restrict__ zt, // (NB, DZZ, NG) f16
    float* __restrict__ out)         // (NB, NTT, DZZ*NKK)
{
  const int lane = threadIdx.x & 31;
  const int wave = threadIdx.x >> 5;        // wave32
  const int pair = blockIdx.x >> 5;         // 0..7 : (b,k)
  const int chunk = blockIdx.x & 31;
  const int tile = chunk * 4 + wave;        // 0..127 M-tiles
  const int b = pair >> 1;
  const int k = pair & 1;
  const int row  = lane & 15;
  const int half = lane >> 4;
  const int m0 = tile * 16;

  // lengthscales: ls = 1e-5 + softplus(p); fold exp(-0.5 x) = exp2(s*x)
  const float p0 = lsp[0 * NKK + k];
  const float p1 = lsp[1 * NKK + k];
  const float ls0 = 1e-5f + (p0 > 20.f ? p0 : log1pf(expf(p0)));
  const float ls1 = 1e-5f + (p1 > 20.f ? p1 : log1pf(expf(p1)));
  const float L2E = 1.4426950408889634f;
  const float s0 = -0.5f * L2E / (ls0 * ls0);
  const float s1 = -0.5f * L2E / (ls1 * ls1);

  // This lane's target point (A-matrix row m0+row)
  const float xt0 = xt[(b * NTT + m0 + row) * 2 + 0];
  const float xt1 = xt[(b * NTT + m0 + row) * 2 + 1];

  const v4f* __restrict__ xg4 = (const v4f*)(xg + (size_t)b * NG * 2);

  // Per-N-tile row offsets into transposed Z (this lane's column = row)
  int zoff[8];
#pragma unroll
  for (int n = 0; n < 8; ++n) zoff[n] = (b * DZZ + n * 16 + row) * NG;

  v8f acc[8] = {};

  for (int g0 = 0; g0 < NG; g0 += 32) {
    // ----- A operand: 16 RBF weights in exact 16-bit A layout -----
    // element e = r*8+i  ->  K = 16*r + 8*half + i
    v16h a;
    const int gA = g0 + 8 * half;
#pragma unroll
    for (int r = 0; r < 2; ++r) {
      const int gr = gA + 16 * r;
#pragma unroll
      for (int i2 = 0; i2 < 4; ++i2) {         // two grid points per float4
        const v4f x = xg4[(gr >> 1) + i2];
        float d0 = xt0 - x.x, d1 = xt1 - x.y;
        float e0 = fmaf(d1 * s1, d1, (d0 * s0) * d0);
        float w0 = __builtin_amdgcn_exp2f(e0);
        d0 = xt0 - x.z; d1 = xt1 - x.w;
        float e1 = fmaf(d1 * s1, d1, (d0 * s0) * d0);
        float w1 = __builtin_amdgcn_exp2f(e1);
        a[r * 8 + 2 * i2 + 0] = (_Float16)w0;
        a[r * 8 + 2 * i2 + 1] = (_Float16)w1;
      }
    }

    // ----- B operands: two contiguous 16B loads per N-tile (pre-transposed Z)
#pragma unroll
    for (int n = 0; n < 8; ++n) {
      const v8h lo = *(const v8h*)(zt + zoff[n] + g0 + 8 * half);
      const v8h hi = *(const v8h*)(zt + zoff[n] + g0 + 16 + 8 * half);
      const v16h bm = __builtin_shufflevector(lo, hi,
          0, 1, 2, 3, 4, 5, 6, 7, 8, 9, 10, 11, 12, 13, 14, 15);
      acc[n] = __builtin_amdgcn_wmma_f32_16x16x32_f16(
          /*neg_a=*/false, a, /*neg_b=*/false, bm,
          /*c_mod=*/(short)0, acc[n], /*reuse_a=*/false, /*reuse_b=*/false);
    }
  }

  // ----- Epilogue: C layout VGPR r -> M = r + 8*half, N = lane&15 -----
#pragma unroll
  for (int n = 0; n < 8; ++n) {
#pragma unroll
    for (int r = 0; r < 8; ++r) {
      const int t = m0 + r + 8 * half;
      out[((b * NTT + t) * (DZZ * NKK)) + (n * 16 + row) * NKK + k] = acc[n][r];
    }
  }
}

// ---------------------------------------------------------------------------
extern "C" void kernel_launch(void* const* d_in, const int* in_sizes, int n_in,
                              void* d_out, int out_size, void* d_ws, size_t ws_size,
                              hipStream_t stream) {
  const float* x_grid = (const float*)d_in[0];   // (4,64,64,2)  -> (4,4096,2)
  const float* z_grid = (const float*)d_in[1];   // (4,64,64,128)-> (4,4096,128)
  const float* xt     = (const float*)d_in[2];   // (4,2048,2)
  const float* lsp    = (const float*)d_in[3];   // (2,2)
  float* out = (float*)d_out;                    // (4,2048,256)
  _Float16* zt = (_Float16*)d_ws;                // 4 MB f16 scratch

  // 1) cast+transpose Z into WMMA-friendly (b, z, g) f16 layout
  transpose_cast_z<<<dim3(NB * (NG / 32) * (DZZ / 32)), dim3(256), 0, stream>>>(
      z_grid, zt);
  // 2) fused RBF-weight generation + f16 WMMA GEMM (f32 accumulate)
  setconv_wmma<<<dim3(256), dim3(128), 0, stream>>>(x_grid, xt, lsp, zt, out);
}
